// MultiscaleSAttn_41961830482666
// MI455X (gfx1250) — matrix operand order repaired and checked by reading, not compiled, not run on hardware
//
#include <hip/hip_runtime.h>
#include <hip/hip_bf16.h>
#include <math.h>

#define NODES   512
#define BATCH   16
#define SCALE_S 512
#define SCALE_M 1024
#define SCALE_L 2048
#define SEQTOT  (SCALE_S + SCALE_M + SCALE_L)   // 3584
#define LN_EPS  1e-5f

#define AS1 __attribute__((address_space(1)))
#define AS3 __attribute__((address_space(3)))

#if __has_builtin(__builtin_amdgcn_global_load_async_to_lds_b128) && \
    __has_builtin(__builtin_amdgcn_s_wait_asynccnt)
#define HAVE_ASYNC_LDS 1
#endif

typedef __attribute__((ext_vector_type(16))) __bf16 v16bf;
typedef __attribute__((ext_vector_type(8)))  float  v8f;
typedef int v4i __attribute__((vector_size(16)));   // matches builtin param type

struct Frag8 { unsigned int w[8]; };   // 32 bytes = one WMMA 16-bit operand per lane

__device__ __forceinline__ unsigned short f2bf(float f) {
  unsigned int u = __builtin_bit_cast(unsigned int, f);
  u += 0x7fffu + ((u >> 16) & 1u);     // round-to-nearest-even
  return (unsigned short)(u >> 16);
}

__device__ __forceinline__ v8f wmma_bf16(const Frag8& a, const Frag8& b, v8f c) {
  return __builtin_amdgcn_wmma_f32_16x16x32_bf16(
      false, __builtin_bit_cast(v16bf, a),
      false, __builtin_bit_cast(v16bf, b),
      (short)0, c, false, false);
}

// A-fragment 16(M)x32(K) bf16, row-major source (lane&15 = M row).
// Lanes 0-15: K 0-7 in w0-3, K 16-23 in w4-7; lanes 16-31: K 8-15 / 24-31.
__device__ __forceinline__ Frag8 load_fragA(const unsigned short* __restrict__ src,
                                            int ld, int lane) {
  const int half = lane >> 4;
  const int r    = lane & 15;
  const unsigned short* p = src + (size_t)r * ld + half * 8;
  uint4 lo = *reinterpret_cast<const uint4*>(p);        // K k0..k0+7
  uint4 hi = *reinterpret_cast<const uint4*>(p + 16);   // K k0+16..k0+23
  Frag8 f;
  f.w[0]=lo.x; f.w[1]=lo.y; f.w[2]=lo.z; f.w[3]=lo.w;
  f.w[4]=hi.x; f.w[5]=hi.y; f.w[6]=hi.z; f.w[7]=hi.w;
  return f;
}

// B-fragment 32(K)x16(N) bf16 from the K-contiguous (transposed) view:
// src row-major (N rows, K cols). lane&15 = N column; lanes 0-15 hold K 0-7/16-23,
// lanes 16-31 hold K 8-15/24-31 (same K order as load_fragA).
__device__ __forceinline__ Frag8 load_fragB(const unsigned short* __restrict__ src,
                                            int ld, int lane) {
  const int half = lane >> 4;
  const int r    = lane & 15;
  const unsigned short* p = src + (size_t)r * ld + half * 8;
  uint4 lo = *reinterpret_cast<const uint4*>(p);
  uint4 hi = *reinterpret_cast<const uint4*>(p + 16);
  Frag8 f;
  f.w[0]=lo.x; f.w[1]=lo.y; f.w[2]=lo.z; f.w[3]=lo.w;
  f.w[4]=hi.x; f.w[5]=hi.y; f.w[6]=hi.z; f.w[7]=hi.w;
  return f;
}

// ---------------------------------------------------------------- LayerNorm
__global__ __launch_bounds__(256)
void ln_kernel(const float* __restrict__ x, const float* __restrict__ gamma,
               const float* __restrict__ beta, unsigned short* __restrict__ h) {
  __shared__ float s_sum[256];
  __shared__ float s_sq[256];
  const int row = blockIdx.x;
  const int t   = threadIdx.x;
  const float* xr = x + (size_t)row * NODES;
  float a = xr[t], b = xr[t + 256];
  s_sum[t] = a + b;
  s_sq[t]  = a * a + b * b;
  __syncthreads();
  for (int off = 128; off > 0; off >>= 1) {
    if (t < off) { s_sum[t] += s_sum[t + off]; s_sq[t] += s_sq[t + off]; }
    __syncthreads();
  }
  const float mu   = s_sum[0] * (1.0f / NODES);
  const float var  = s_sq[0] * (1.0f / NODES) - mu * mu;
  const float rstd = rsqrtf(var + LN_EPS);
  unsigned short* hr = h + (size_t)row * NODES;
  hr[t]       = f2bf((a - mu) * rstd * gamma[t] + beta[t]);
  hr[t + 256] = f2bf((b - mu) * rstd * gamma[t + 256] + beta[t + 256]);
}

// ------------------------------------------- weight fp32(K,N) -> bf16(N,K)
__global__ void wconv_kernel(const float* __restrict__ W,
                             unsigned short* __restrict__ WT, int K, int N) {
  int idx = blockIdx.x * blockDim.x + threadIdx.x;
  if (idx >= K * N) return;
  int n = idx / K;
  int k = idx - n * K;
  WT[(size_t)n * K + k] = f2bf(W[(size_t)k * N + n]);
}

// ------------------------------------------------------- QKV projection GEMM
// Block: 128 threads (4 waves). Tile: 64 rows x 64 cols. Wave = 16x64.
__global__ __launch_bounds__(128)
void qkv_gemm_kernel(const unsigned short* __restrict__ h,
                     const unsigned short* __restrict__ WT,   // (1536, 512)
                     const float* __restrict__ bias,          // (1536)
                     unsigned short* __restrict__ qbuf,
                     unsigned short* __restrict__ kbuf,
                     unsigned short* __restrict__ vT,
                     int Lseg, int soff) {
  const int lane = threadIdx.x & 31;
  const int wave = threadIdx.x >> 5;
  const int half = lane >> 4;
  const int ln15 = lane & 15;
  const int colb = blockIdx.x * 64;
  const int rb   = Lseg >> 6;
  const int b    = blockIdx.y / rb;
  const int lt   = blockIdx.y - b * rb;
  const int tok0 = lt * 64 + wave * 16;
  const size_t hrow = ((size_t)b * SEQTOT + soff + tok0) * NODES;

  v8f acc[4] = {};
  #pragma unroll
  for (int kc = 0; kc < NODES / 32; ++kc) {
    __builtin_prefetch(h + hrow + (kc + 1) * 32, 0, 3);
    Frag8 af = load_fragA(h + hrow + kc * 32, NODES, lane);
    Frag8 bfr[4];
    #pragma unroll
    for (int t = 0; t < 4; ++t)
      bfr[t] = load_fragB(WT + (size_t)(colb + t * 16) * NODES + kc * 32, NODES, lane);
    #pragma unroll
    for (int t = 0; t < 4; ++t)
      acc[t] = wmma_bf16(af, bfr[t], acc[t]);
  }
  if (colb < 2 * NODES) {                         // q or k: row-major bf16
    unsigned short* dst = (colb < NODES) ? qbuf : kbuf;
    const int cb = (colb < NODES) ? colb : colb - NODES;
    #pragma unroll
    for (int t = 0; t < 4; ++t) {
      const int   n   = cb + t * 16 + ln15;
      const float bia = bias[colb + t * 16 + ln15];
      #pragma unroll
      for (int j = 0; j < 8; ++j) {
        const int m = j + 8 * half;
        dst[((size_t)b * SEQTOT + soff + tok0 + m) * NODES + n] = f2bf(acc[t][j] + bia);
      }
    }
  } else {                                        // v: store transposed, packed
    const size_t vbase = ((size_t)b * SEQTOT + soff) * NODES;
    #pragma unroll
    for (int t = 0; t < 4; ++t) {
      const int   d   = (colb - 2 * NODES) + t * 16 + ln15;
      const float bia = bias[colb + t * 16 + ln15];
      unsigned int p0 = (unsigned)f2bf(acc[t][0] + bia) | ((unsigned)f2bf(acc[t][1] + bia) << 16);
      unsigned int p1 = (unsigned)f2bf(acc[t][2] + bia) | ((unsigned)f2bf(acc[t][3] + bia) << 16);
      unsigned int p2 = (unsigned)f2bf(acc[t][4] + bia) | ((unsigned)f2bf(acc[t][5] + bia) << 16);
      unsigned int p3 = (unsigned)f2bf(acc[t][6] + bia) | ((unsigned)f2bf(acc[t][7] + bia) << 16);
      uint4 vv; vv.x = p0; vv.y = p1; vv.z = p2; vv.w = p3;
      *reinterpret_cast<uint4*>(vT + vbase + (size_t)d * Lseg + tok0 + 8 * half) = vv;
    }
  }
}

// ------------------------------------------------- flash-style attention
// Block: 128 threads (4 waves), 16 query rows. Key tiles of 128.
// q tile (16x512 bf16 = 16KB) staged in LDS (async copy when available).
__global__ __launch_bounds__(128)
void attn_kernel(const unsigned short* __restrict__ qbuf,
                 const unsigned short* __restrict__ kbuf,
                 const unsigned short* __restrict__ vT,
                 unsigned short* __restrict__ aout,
                 int Lseg, int soff, float scaling) {
  __shared__ __align__(16) unsigned short sQ[16 * NODES];   // 16 KB
  __shared__ __align__(16) float          sS[16 * 128];     //  8 KB
  __shared__ __align__(16) unsigned short sP[16 * 128];     //  4 KB
  __shared__ float sScale[16];
  __shared__ float sInv[16];
  const int lane = threadIdx.x & 31;
  const int wave = threadIdx.x >> 5;
  const int half = lane >> 4;
  const int ln15 = lane & 15;
  const int qtiles = Lseg >> 4;
  const int b  = blockIdx.x / qtiles;
  const int qt = blockIdx.x - b * qtiles;
  const size_t segbase = ((size_t)b * SEQTOT + soff) * NODES;
  const size_t qrow    = segbase + (size_t)qt * 16 * NODES;

  // ---- stage the contiguous 16x512 q tile into LDS ----
#ifdef HAVE_ASYNC_LDS
  {
    const unsigned short* g = qbuf + qrow;
    #pragma unroll
    for (int it = 0; it < 8; ++it) {
      __builtin_amdgcn_global_load_async_to_lds_b128(
          (AS1 v4i*)(g + (size_t)it * 1024 + threadIdx.x * 8),
          (AS3 v4i*)(sQ + (size_t)it * 1024 + threadIdx.x * 8), 0, 0);
    }
    __builtin_amdgcn_s_wait_asynccnt(0);
  }
#else
  {
    const uint4* gsrc = reinterpret_cast<const uint4*>(qbuf + qrow);
    uint4* ldst = reinterpret_cast<uint4*>(sQ);
    #pragma unroll
    for (int it = 0; it < 8; ++it)
      ldst[threadIdx.x + it * 128] = gsrc[threadIdx.x + it * 128];
  }
#endif
  __syncthreads();

  v8f acc[8] = {};
  float m_run = -INFINITY, l_run = 0.0f;

  const int nkt = Lseg >> 7;
  for (int kt = 0; kt < nkt; ++kt) {
    // Phase A: this wave computes scores for 32 keys (2 n-tiles of 16).
    {
      v8f s0 = {}, s1 = {};
      const size_t krow0 = segbase + ((size_t)kt * 128 + wave * 32) * NODES;
      const size_t krow1 = krow0 + 16 * NODES;
      __builtin_prefetch(kbuf + krow0 + 128 * NODES, 0, 3);
      #pragma unroll
      for (int kc = 0; kc < 16; ++kc) {
        Frag8 af = load_fragA(sQ + kc * 32, NODES, lane);
        Frag8 b0 = load_fragB(kbuf + krow0 + kc * 32, NODES, lane);
        Frag8 b1 = load_fragB(kbuf + krow1 + kc * 32, NODES, lane);
        s0 = wmma_bf16(af, b0, s0);
        s1 = wmma_bf16(af, b1, s1);
      }
      #pragma unroll
      for (int j = 0; j < 8; ++j) {
        sS[(j + 8 * half) * 128 + wave * 32 + ln15]      = s0[j] * scaling;
        sS[(j + 8 * half) * 128 + wave * 32 + 16 + ln15] = s1[j] * scaling;
      }
    }
    __syncthreads();
    // Phase B: online softmax, one thread per query row.
    if (threadIdx.x < 16) {
      const int r = threadIdx.x;
      float mx = m_run;
      for (int i = 0; i < 128; ++i) mx = fmaxf(mx, sS[r * 128 + i]);
      const float scale = __expf(m_run - mx);
      float sum = 0.0f;
      for (int i = 0; i < 128; ++i) {
        const float p = __expf(sS[r * 128 + i] - mx);
        sP[r * 128 + i] = f2bf(p);
        sum += p;
      }
      l_run = l_run * scale + sum;
      m_run = mx;
      sScale[r] = scale;
    }
    __syncthreads();
    // Phase C: rescale accumulators, accumulate P @ V over this wave's 128 dims.
    float rs[8];
    #pragma unroll
    for (int j = 0; j < 8; ++j) rs[j] = sScale[j + 8 * half];
    #pragma unroll
    for (int t = 0; t < 8; ++t)
      #pragma unroll
      for (int j = 0; j < 8; ++j) acc[t][j] *= rs[j];
    #pragma unroll
    for (int kc = 0; kc < 4; ++kc) {
      Frag8 pf = load_fragA(sP + kc * 32, 128, lane);
      const size_t kcol = (size_t)kt * 128 + kc * 32;
      #pragma unroll
      for (int t = 0; t < 8; t += 2) {
        const int d0 = wave * 128 + t * 16;
        Frag8 b0 = load_fragB(vT + segbase + (size_t)d0 * Lseg + kcol, Lseg, lane);
        Frag8 b1 = load_fragB(vT + segbase + (size_t)(d0 + 16) * Lseg + kcol, Lseg, lane);
        acc[t]     = wmma_bf16(pf, b0, acc[t]);
        acc[t + 1] = wmma_bf16(pf, b1, acc[t + 1]);
      }
    }
    __syncthreads();
  }
  if (threadIdx.x < 16) sInv[threadIdx.x] = 1.0f / l_run;
  __syncthreads();
  float riv[8];
  #pragma unroll
  for (int j = 0; j < 8; ++j) riv[j] = sInv[j + 8 * half];
  #pragma unroll
  for (int t = 0; t < 8; ++t) {
    const int n = wave * 128 + t * 16 + ln15;
    #pragma unroll
    for (int j = 0; j < 8; ++j) {
      const int m = j + 8 * half;
      aout[segbase + ((size_t)qt * 16 + m) * NODES + n] = f2bf(acc[t][j] * riv[j]);
    }
  }
}

// --------------------------------------- output projection + bias + residual
__global__ __launch_bounds__(128)
void out_gemm_kernel(const unsigned short* __restrict__ aout,
                     const unsigned short* __restrict__ WT,   // (512, 512)
                     const float* __restrict__ bout,
                     const float* __restrict__ x,
                     float* __restrict__ out) {
  const int lane = threadIdx.x & 31;
  const int wave = threadIdx.x >> 5;
  const int half = lane >> 4;
  const int ln15 = lane & 15;
  const int colb = blockIdx.x * 64;
  const size_t row0 = (size_t)blockIdx.y * 64 + wave * 16;
  v8f acc[4] = {};
  #pragma unroll
  for (int kc = 0; kc < NODES / 32; ++kc) {
    __builtin_prefetch(aout + row0 * NODES + (kc + 1) * 32, 0, 3);
    Frag8 af = load_fragA(aout + row0 * NODES + kc * 32, NODES, lane);
    Frag8 bfr[4];
    #pragma unroll
    for (int t = 0; t < 4; ++t)
      bfr[t] = load_fragB(WT + (size_t)(colb + t * 16) * NODES + kc * 32, NODES, lane);
    #pragma unroll
    for (int t = 0; t < 4; ++t)
      acc[t] = wmma_bf16(af, bfr[t], acc[t]);
  }
  #pragma unroll
  for (int t = 0; t < 4; ++t) {
    const int   n   = colb + t * 16 + ln15;
    const float bia = bout[n];
    #pragma unroll
    for (int j = 0; j < 8; ++j) {
      const size_t r = row0 + j + 8 * half;
      out[r * NODES + n] = acc[t][j] + bia + x[r * NODES + n];
    }
  }
}

extern "C" void kernel_launch(void* const* d_in, const int* in_sizes, int n_in,
                              void* d_out, int out_size, void* d_ws, size_t ws_size,
                              hipStream_t stream) {
  (void)in_sizes; (void)n_in; (void)out_size; (void)ws_size;
  const float* x     = (const float*)d_in[0];
  const float* gamma = (const float*)d_in[1];
  const float* beta  = (const float*)d_in[2];
  const float* Wq_s  = (const float*)d_in[3];
  const float* bq_s  = (const float*)d_in[4];
  const float* Wq_m  = (const float*)d_in[5];
  const float* bq_m  = (const float*)d_in[6];
  const float* Wq_l  = (const float*)d_in[7];
  const float* bq_l  = (const float*)d_in[8];
  const float* Wout  = (const float*)d_in[9];
  const float* bout  = (const float*)d_in[10];
  float* out = (float*)d_out;

  const size_t H = (size_t)BATCH * SEQTOT * NODES;       // activation elements
  unsigned short* ws    = (unsigned short*)d_ws;
  unsigned short* h_bf  = ws;                            // H (bf16) — reused as aout
  unsigned short* q_bf  = ws + H;
  unsigned short* k_bf  = ws + 2 * H;
  unsigned short* vT_bf = ws + 3 * H;
  unsigned short* wqT_s = ws + 4 * H;                    // 1536*512
  unsigned short* wqT_m = wqT_s + (size_t)1536 * 512;
  unsigned short* wqT_l = wqT_m + (size_t)1536 * 512;
  unsigned short* woT   = wqT_l + (size_t)1536 * 512;    // 512*512
  unsigned short* ao_bf = h_bf;  // h dead after QKV GEMMs -> alias for attn out

  ln_kernel<<<BATCH * SEQTOT, 256, 0, stream>>>(x, gamma, beta, h_bf);

  const int wq = 1536 * 512, wo = 512 * 512;
  wconv_kernel<<<(wq + 255) / 256, 256, 0, stream>>>(Wq_s, wqT_s, 512, 1536);
  wconv_kernel<<<(wq + 255) / 256, 256, 0, stream>>>(Wq_m, wqT_m, 512, 1536);
  wconv_kernel<<<(wq + 255) / 256, 256, 0, stream>>>(Wq_l, wqT_l, 512, 1536);
  wconv_kernel<<<(wo + 255) / 256, 256, 0, stream>>>(Wout, woT, 512, 512);

  qkv_gemm_kernel<<<dim3(24, BATCH * (SCALE_S / 64)), 128, 0, stream>>>(
      h_bf, wqT_s, bq_s, q_bf, k_bf, vT_bf, SCALE_S, 0);
  qkv_gemm_kernel<<<dim3(24, BATCH * (SCALE_M / 64)), 128, 0, stream>>>(
      h_bf, wqT_m, bq_m, q_bf, k_bf, vT_bf, SCALE_M, SCALE_S);
  qkv_gemm_kernel<<<dim3(24, BATCH * (SCALE_L / 64)), 128, 0, stream>>>(
      h_bf, wqT_l, bq_l, q_bf, k_bf, vT_bf, SCALE_L, SCALE_S + SCALE_M);

  const float scaling = 0.044194173824159216f;  // 512^-0.5
  attn_kernel<<<BATCH * (SCALE_S / 16), 128, 0, stream>>>(
      q_bf, k_bf, vT_bf, ao_bf, SCALE_S, 0, scaling);
  attn_kernel<<<BATCH * (SCALE_M / 16), 128, 0, stream>>>(
      q_bf, k_bf, vT_bf, ao_bf, SCALE_M, SCALE_S, scaling);
  attn_kernel<<<BATCH * (SCALE_L / 16), 128, 0, stream>>>(
      q_bf, k_bf, vT_bf, ao_bf, SCALE_L, SCALE_S + SCALE_M, scaling);

  out_gemm_kernel<<<dim3(8, BATCH * SEQTOT / 64), 128, 0, stream>>>(
      ao_bf, woT, bout, x, out);
}
